// MultiHeadAttention_3985729651423
// MI455X (gfx1250) — matrix operand order repaired and checked by reading, not compiled, not run on hardware
//
#include <hip/hip_runtime.h>
#include <hip/hip_bf16.h>
#include <math.h>

// ---------------------------------------------------------------------------
// MHA forward for MI455X (gfx1250, wave32, WMMA).
// d_out = [ output (S*768 f32) | attn_weights (12*S*S f32) ]
// Each wave computes a 32x64 output tile: 2 A-fragments share 4 B-fragments,
// giving 8 independent v_wmma per k-step (hides the WMMA->WMMA RAW hazard and
// halves B-fragment vmem traffic vs a 16x64 tile).
// ---------------------------------------------------------------------------

#define D_MODEL 768
#define NUM_HEADS 12
#define D_K 64
#define SEQ 4096

typedef __attribute__((ext_vector_type(16))) _Float16 v16h;
typedef __attribute__((ext_vector_type(8)))  _Float16 v8h;
typedef __attribute__((ext_vector_type(8)))  float    v8f;
typedef __attribute__((ext_vector_type(4)))  float    v4f;

// ---- fragment loaders (wave32 WMMA 16x16x32 f16 layouts, per ISA 7.12.2) ----

// A fragment (16x32, f16 source, row-major, stride lda halves).
// lane L (L<16): row M=L, halves K=0..7 then K=16..23
// lane L (L>=16): row M=L-16, halves K=8..15 then K=24..31
__device__ __forceinline__ v16h load_a_f16(const _Float16* A, int lda, int lane) {
    int r = lane & 15;
    int hs = (lane >> 4) << 3;         // 0 or 8
    const _Float16* p = A + (size_t)r * lda + hs;
    v8h lo = *(const v8h*)(p);
    v8h hi = *(const v8h*)(p + 16);
    return __builtin_shufflevector(lo, hi, 0,1,2,3,4,5,6,7,8,9,10,11,12,13,14,15);
}

// A fragment from f32 source with in-register convert to f16.
__device__ __forceinline__ v16h load_a_f32(const float* A, int lda, int lane) {
    int r = lane & 15;
    int hs = (lane >> 4) << 3;
    const float* p = A + (size_t)r * lda + hs;
    v4f a0 = *(const v4f*)(p);
    v4f a1 = *(const v4f*)(p + 4);
    v4f a2 = *(const v4f*)(p + 16);
    v4f a3 = *(const v4f*)(p + 20);
    v16h out;
#pragma unroll
    for (int i = 0; i < 4; ++i) {
        out[i]      = (_Float16)a0[i];
        out[4 + i]  = (_Float16)a1[i];
        out[8 + i]  = (_Float16)a2[i];
        out[12 + i] = (_Float16)a3[i];
    }
    return out;
}

// B fragment (32x16) from Bt stored N-major: Bt[n][k], stride ldbt halves.
// lane n (n<16): column n, K=0..7 then K=16..23 ; lane n+16: column n, K=8..15 then K=24..31.
__device__ __forceinline__ v16h load_b_nk(const _Float16* Bt, int ldbt, int lane) {
    int n  = lane & 15;
    int kh = (lane >> 4) << 3;         // 0 or 8
    const _Float16* p = Bt + (size_t)n * ldbt + kh;
    v8h lo = *(const v8h*)(p);
    v8h hi = *(const v8h*)(p + 16);
    return __builtin_shufflevector(lo, hi, 0,1,2,3,4,5,6,7,8,9,10,11,12,13,14,15);
}

__device__ __forceinline__ v8f wmma_f16(v16h a, v16h b, v8f c) {
    return __builtin_amdgcn_wmma_f32_16x16x32_f16(false, a, false, b,
                                                  (short)0, c, false, false);
}

// ---------------------------------------------------------------------------
// f32 -> f16 elementwise (weights)
// ---------------------------------------------------------------------------
__global__ void mha_cvt_f16(const float* __restrict__ src,
                            _Float16* __restrict__ dst, int n) {
    for (int i = blockIdx.x * blockDim.x + threadIdx.x; i < n;
         i += gridDim.x * blockDim.x)
        dst[i] = (_Float16)src[i];
}

// ---------------------------------------------------------------------------
// Projection: out = X(f32, MxK) @ Wt^T + bias (Wt is the torch weight,
// N x K row-major == Bt operand directly).  out f16, optionally transposed
// (out[n*M+m]) so V can be written as vT[H][64][S].
// grid: (N/64, M/128), block 128 (4 waves, each 32x64 tile).
// ---------------------------------------------------------------------------
__global__ void mha_proj_f16(const float* __restrict__ X,
                             const _Float16* __restrict__ Wt,
                             const float* __restrict__ bias,
                             _Float16* __restrict__ out,
                             int M, int K, int N, int transpose_out) {
    const int lane = threadIdx.x & 31;
    const int wave = threadIdx.x >> 5;
    const int m0 = (blockIdx.y * 4 + wave) * 32;
    const int n0 = blockIdx.x * 64;

    v8f acc0[4] = {};
    v8f acc1[4] = {};
    for (int k = 0; k < K; k += 32) {
        v16h a0 = load_a_f32(X + (size_t)m0 * K + k, K, lane);
        v16h a1 = load_a_f32(X + (size_t)(m0 + 16) * K + k, K, lane);
#pragma unroll
        for (int j = 0; j < 4; ++j) {
            v16h b = load_b_nk(Wt + (size_t)(n0 + j * 16) * K + k, K, lane);
            acc0[j] = wmma_f16(a0, b, acc0[j]);
            acc1[j] = wmma_f16(a1, b, acc1[j]);
        }
    }

    const int nl = lane & 15;
    const int mb = (lane >> 4) * 8;
#pragma unroll
    for (int j = 0; j < 4; ++j) {
        int n = n0 + j * 16 + nl;
        float bv = bias ? bias[n] : 0.0f;
#pragma unroll
        for (int i = 0; i < 8; ++i) {
            int ma = m0 + mb + i;
            int mc = m0 + 16 + mb + i;
            float va = acc0[j][i] + bv;
            float vc = acc1[j][i] + bv;
            if (transpose_out) {
                out[(size_t)n * M + ma] = (_Float16)va;
                out[(size_t)n * M + mc] = (_Float16)vc;
            } else {
                out[(size_t)ma * N + n] = (_Float16)va;
                out[(size_t)mc * N + n] = (_Float16)vc;
            }
        }
    }
}

// ---------------------------------------------------------------------------
// Scores: attn[h] = (q[h] @ k[h]^T) / sqrt(d_k).  q,k stored [S][768] f16;
// head h occupies columns h*64..h*64+63 -> q rows are the A operand and k
// rows are the Bt operand directly.  K-dim = 64 -> 2 wmma k-steps.
// grid: (S/64, S/128, H), block 128 (each wave 32x64).
// ---------------------------------------------------------------------------
__global__ void mha_scores(const _Float16* __restrict__ q,
                           const _Float16* __restrict__ k,
                           float* __restrict__ attn, int S) {
    const int lane = threadIdx.x & 31;
    const int wave = threadIdx.x >> 5;
    const int h  = blockIdx.z;
    const int m0 = (blockIdx.y * 4 + wave) * 32;
    const int n0 = blockIdx.x * 64;

    const _Float16* qh = q + h * D_K;
    const _Float16* kh = k + h * D_K;

    v8f acc0[4] = {};
    v8f acc1[4] = {};
#pragma unroll
    for (int kk = 0; kk < D_K; kk += 32) {
        v16h a0 = load_a_f16(qh + (size_t)m0 * D_MODEL + kk, D_MODEL, lane);
        v16h a1 = load_a_f16(qh + (size_t)(m0 + 16) * D_MODEL + kk, D_MODEL, lane);
#pragma unroll
        for (int j = 0; j < 4; ++j) {
            v16h b = load_b_nk(kh + (size_t)(n0 + j * 16) * D_MODEL + kk,
                               D_MODEL, lane);
            acc0[j] = wmma_f16(a0, b, acc0[j]);
            acc1[j] = wmma_f16(a1, b, acc1[j]);
        }
    }

    const float scale = 0.125f;  // 1/sqrt(64)
    float* ah = attn + (size_t)h * S * S;
    const int nl = lane & 15;
    const int mb = (lane >> 4) * 8;
#pragma unroll
    for (int j = 0; j < 4; ++j) {
        int n = n0 + j * 16 + nl;
#pragma unroll
        for (int i = 0; i < 8; ++i) {
            ah[(size_t)(m0 + mb + i) * S + n]      = acc0[j][i] * scale;
            ah[(size_t)(m0 + 16 + mb + i) * S + n] = acc1[j][i] * scale;
        }
    }
}

// ---------------------------------------------------------------------------
// Row softmax in place; one 256-thread block per row of 4096.
// exp values held in registers -> single store pass.
// ---------------------------------------------------------------------------
__global__ void mha_softmax(float* __restrict__ attn, int S) {
    float* p = attn + (size_t)blockIdx.x * S;
    const int t = threadIdx.x;
    const int PER = SEQ / 256;  // 16

    float loc[PER];
    float m = -3.402823466e38f;
#pragma unroll
    for (int i = 0; i < PER; ++i) {
        loc[i] = p[t + i * 256];
        m = fmaxf(m, loc[i]);
    }

    __shared__ float red[256];
    red[t] = m;
    __syncthreads();
    for (int s = 128; s > 0; s >>= 1) {
        if (t < s) red[t] = fmaxf(red[t], red[t + s]);
        __syncthreads();
    }
    m = red[0];
    __syncthreads();

    float sum = 0.0f;
#pragma unroll
    for (int i = 0; i < PER; ++i) {
        loc[i] = __expf(loc[i] - m);
        sum += loc[i];
    }
    red[t] = sum;
    __syncthreads();
    for (int s = 128; s > 0; s >>= 1) {
        if (t < s) red[t] += red[t + s];
        __syncthreads();
    }
    float inv = 1.0f / red[0];
#pragma unroll
    for (int i = 0; i < PER; ++i) p[t + i * 256] = loc[i] * inv;
}

// ---------------------------------------------------------------------------
// ctx = attn @ v.  attn f32 (converted per-fragment), vT[H][64][S] as Bt.
// Each wave: 32 query rows x all 64 head-dims; K loop over S (128 steps).
// grid: (S/128, 1, H), block 128.  ctx written f16 [S][768] (heads merged).
// ---------------------------------------------------------------------------
__global__ void mha_ctx(const float* __restrict__ attn,
                        const _Float16* __restrict__ vT,
                        _Float16* __restrict__ ctx, int S) {
    const int lane = threadIdx.x & 31;
    const int wave = threadIdx.x >> 5;
    const int h  = blockIdx.z;
    const int m0 = (blockIdx.x * 4 + wave) * 32;

    const float*    A = attn + (size_t)h * S * S + (size_t)m0 * S;
    const _Float16* B = vT + (size_t)h * D_K * S;

    v8f acc0[4] = {};
    v8f acc1[4] = {};
    for (int kk = 0; kk < S; kk += 32) {
        __builtin_prefetch(A + kk + 256, 0, 0);   // global_prefetch_b8
        v16h a0 = load_a_f32(A + kk, S, lane);
        v16h a1 = load_a_f32(A + (size_t)16 * S + kk, S, lane);
#pragma unroll
        for (int j = 0; j < 4; ++j) {
            v16h b = load_b_nk(B + (size_t)(j * 16) * S + kk, S, lane);
            acc0[j] = wmma_f16(a0, b, acc0[j]);
            acc1[j] = wmma_f16(a1, b, acc1[j]);
        }
    }

    const int nl = lane & 15;
    const int mb = (lane >> 4) * 8;
#pragma unroll
    for (int j = 0; j < 4; ++j) {
        int d = j * 16 + nl;
#pragma unroll
        for (int i = 0; i < 8; ++i) {
            ctx[(size_t)(m0 + mb + i) * D_MODEL + h * D_K + d] =
                (_Float16)acc0[j][i];
            ctx[(size_t)(m0 + 16 + mb + i) * D_MODEL + h * D_K + d] =
                (_Float16)acc1[j][i];
        }
    }
}

// ---------------------------------------------------------------------------
// Output projection: out(f32) = ctx(f16) @ Wo^T + bo.
// grid: (N/64, M/128), block 128.
// ---------------------------------------------------------------------------
__global__ void mha_outproj(const _Float16* __restrict__ ctx,
                            const _Float16* __restrict__ Wo,
                            const float* __restrict__ bo,
                            float* __restrict__ out, int M, int K, int N) {
    const int lane = threadIdx.x & 31;
    const int wave = threadIdx.x >> 5;
    const int m0 = (blockIdx.y * 4 + wave) * 32;
    const int n0 = blockIdx.x * 64;

    v8f acc0[4] = {};
    v8f acc1[4] = {};
    for (int k = 0; k < K; k += 32) {
        v16h a0 = load_a_f16(ctx + (size_t)m0 * K + k, K, lane);
        v16h a1 = load_a_f16(ctx + (size_t)(m0 + 16) * K + k, K, lane);
#pragma unroll
        for (int j = 0; j < 4; ++j) {
            v16h b = load_b_nk(Wo + (size_t)(n0 + j * 16) * K + k, K, lane);
            acc0[j] = wmma_f16(a0, b, acc0[j]);
            acc1[j] = wmma_f16(a1, b, acc1[j]);
        }
    }

    const int nl = lane & 15;
    const int mb = (lane >> 4) * 8;
#pragma unroll
    for (int j = 0; j < 4; ++j) {
        int n = n0 + j * 16 + nl;
        float bv = bo ? bo[n] : 0.0f;
#pragma unroll
        for (int i = 0; i < 8; ++i) {
            out[(size_t)(m0 + mb + i) * N + n]      = acc0[j][i] + bv;
            out[(size_t)(m0 + 16 + mb + i) * N + n] = acc1[j][i] + bv;
        }
    }
}

// ---------------------------------------------------------------------------
extern "C" void kernel_launch(void* const* d_in, const int* in_sizes, int n_in,
                              void* d_out, int out_size, void* d_ws, size_t ws_size,
                              hipStream_t stream) {
    const float* Q  = (const float*)d_in[0];
    const float* K  = (const float*)d_in[1];
    const float* V  = (const float*)d_in[2];
    const float* Wq = (const float*)d_in[3];
    const float* bq = (const float*)d_in[4];
    const float* Wk = (const float*)d_in[5];
    const float* bk = (const float*)d_in[6];
    const float* Wv = (const float*)d_in[7];
    const float* bv = (const float*)d_in[8];
    const float* Wo = (const float*)d_in[9];
    const float* bo = (const float*)d_in[10];

    float* outp = (float*)d_out;                       // S*768 f32
    float* attn = outp + (size_t)SEQ * D_MODEL;        // 12*S*S f32

    // workspace layout (f16)
    const size_t WSZ  = (size_t)D_MODEL * D_MODEL;     // 589824
    const size_t XSZ  = (size_t)SEQ * D_MODEL;         // 3145728
    _Float16* ws   = (_Float16*)d_ws;
    _Float16* Wqh  = ws;            // [768][768]
    _Float16* Wkh  = Wqh + WSZ;
    _Float16* Wvh  = Wkh + WSZ;
    _Float16* Woh  = Wvh + WSZ;
    _Float16* qprj = Woh + WSZ;     // [S][768]
    _Float16* kprj = qprj + XSZ;    // [S][768]
    _Float16* vT   = kprj + XSZ;    // [768][S]  (== [H][64][S])
    _Float16* ctx  = vT   + XSZ;    // [S][768]

    // 1) convert weights to f16
    {
        dim3 b(256), g(576);
        mha_cvt_f16<<<g, b, 0, stream>>>(Wq, Wqh, (int)WSZ);
        mha_cvt_f16<<<g, b, 0, stream>>>(Wk, Wkh, (int)WSZ);
        mha_cvt_f16<<<g, b, 0, stream>>>(Wv, Wvh, (int)WSZ);
        mha_cvt_f16<<<g, b, 0, stream>>>(Wo, Woh, (int)WSZ);
    }

    // 2) Q/K/V projections
    {
        dim3 b(128), g(D_MODEL / 64, SEQ / 128);
        mha_proj_f16<<<g, b, 0, stream>>>(Q, Wqh, bq, qprj, SEQ, D_MODEL, D_MODEL, 0);
        mha_proj_f16<<<g, b, 0, stream>>>(K, Wkh, bk, kprj, SEQ, D_MODEL, D_MODEL, 0);
        mha_proj_f16<<<g, b, 0, stream>>>(V, Wvh, bv, vT,   SEQ, D_MODEL, D_MODEL, 1);
    }

    // 3) scores -> attn region of d_out
    {
        dim3 b(128), g(SEQ / 64, SEQ / 128, NUM_HEADS);
        mha_scores<<<g, b, 0, stream>>>(qprj, kprj, attn, SEQ);
    }

    // 4) softmax rows in place
    {
        dim3 b(256), g(NUM_HEADS * SEQ);
        mha_softmax<<<g, b, 0, stream>>>(attn, SEQ);
    }

    // 5) ctx = attn @ v
    {
        dim3 b(128), g(SEQ / 128, 1, NUM_HEADS);
        mha_ctx<<<g, b, 0, stream>>>(attn, vT, ctx, SEQ);
    }

    // 6) output projection -> d_out
    {
        dim3 b(128), g(D_MODEL / 64, SEQ / 128);
        mha_outproj<<<g, b, 0, stream>>>(ctx, Woh, bo, outp, SEQ, D_MODEL, D_MODEL);
    }
}
